// QTransformer_14293651161097
// MI455X (gfx1250) — matrix operand order repaired and checked by reading, not compiled, not run on hardware
//
#include <hip/hip_runtime.h>
#include <hip/hip_bf16.h>

// ---------------------------------------------------------------------------
// QTransformer forward for MI455X (gfx1250).  All dense projections run on
// v_wmma_f32_16x16x32_f16 (wave32 WMMA) with LDS-staged, double-buffered
// fragments; irregular graph/attention pieces on VALU.
// B=4, A=12, T=80, D=128, H=8, N_DEC=6, E=132.
// ---------------------------------------------------------------------------

typedef __attribute__((ext_vector_type(16))) _Float16 v16h;
typedef __attribute__((ext_vector_type(8)))  _Float16 h8;
typedef __attribute__((ext_vector_type(4)))  _Float16 h4;
typedef __attribute__((ext_vector_type(8)))  float    v8f;

#define ACT_NONE  0
#define ACT_RELU  1
#define ACT_LRELU 2   // slope 0.01 (jax.nn.leaky_relu default)

// problem constants
#define CB 4
#define CA 12
#define CT 80
#define CD 128
#define CG 320          // B*T
#define CROWS 3840      // B*A*T  (also 12*320 decoder rows)
#define CE 132
#define CH 8

// ---------------------------------------------------------------------------
// FAST WMMA GEMM: requires M%32==0, K%32==0, N%64==0, lda%4==0, ldw%4==0.
// Block: 256 threads = 8 waves -> 32x64 tile (2x4 grid of 16x16 WMMA tiles).
// K chunked by 32, double-buffered LDS (1 barrier per chunk), float4 global
// loads, b128/b64 LDS traffic, branch-free epilogue.
// ---------------------------------------------------------------------------
__global__ __launch_bounds__(256)
void gemm_wmma_full(const float* __restrict__ A, int lda,
                    const float* __restrict__ W, int ldw,
                    const float* __restrict__ bias,
                    float* __restrict__ C, int ldc,
                    int M, int N, int K, int act, int accum)
{
    __shared__ __align__(16) _Float16 lA[2][32][40];  // [buf][m][k] 80B row stride
    __shared__ __align__(16) _Float16 lB[2][64][40];  // [buf][n][k] (transposed W)

    int ntN = (N + 63) >> 6;
    int tm = blockIdx.x / ntN;
    int tn = blockIdx.x - tm * ntN;
    int m0 = tm << 5, n0 = tn << 6;

    int tid  = threadIdx.x;
    int lane = tid & 31;
    int wv   = tid >> 5;
    int wr   = wv >> 2;               // 0..1
    int wc   = wv & 3;                // 0..3
    int halfsel = lane >> 4;
    int l15  = lane & 15;

    int ar  = tid >> 3;               // 0..31  A row
    int akq = (tid & 7) << 2;         // 0,4,..,28
    int bk  = tid >> 3;               // 0..31  W k-row
    int bn8 = (tid & 7) << 3;         // 0,8,..,56

    const float* Arow = A + (size_t)(m0 + ar) * lda;
    const float* Wcol = W + n0 + bn8;

    int mL = (wr << 4) + l15;
    int nL = (wc << 4) + l15;

    auto stage = [&](int k0, int bi) {
        float4 av = *(const float4*)(Arow + k0 + akq);
        h4 hv; hv[0] = (_Float16)av.x; hv[1] = (_Float16)av.y;
               hv[2] = (_Float16)av.z; hv[3] = (_Float16)av.w;
        *(h4*)&lA[bi][ar][akq] = hv;                         // ds_store_b64
        const float* Wrow = Wcol + (size_t)(k0 + bk) * ldw;
        float4 w0 = *(const float4*)(Wrow);
        float4 w1 = *(const float4*)(Wrow + 4);
        lB[bi][bn8 + 0][bk] = (_Float16)w0.x;                // transposed stores
        lB[bi][bn8 + 1][bk] = (_Float16)w0.y;
        lB[bi][bn8 + 2][bk] = (_Float16)w0.z;
        lB[bi][bn8 + 3][bk] = (_Float16)w0.w;
        lB[bi][bn8 + 4][bk] = (_Float16)w1.x;
        lB[bi][bn8 + 5][bk] = (_Float16)w1.y;
        lB[bi][bn8 + 6][bk] = (_Float16)w1.z;
        lB[bi][bn8 + 7][bk] = (_Float16)w1.w;
    };

    int nchunks = K >> 5;
    stage(0, 0);                       // prologue into buffer 0
    v8f acc = {};
    for (int c = 0; c < nchunks; ++c) {
        __syncthreads();               // chunk c's stores visible
        if (c + 1 < nchunks) stage((c + 1) << 5, (c + 1) & 1);  // overlap loads
        int bi = c & 1;
        // A 16x32 f16 fragment: elems 0-7 -> K = 8*half + j ; 8-15 -> 16 + 8*half + j
        h8 a0 = *(const h8*)&lA[bi][mL][halfsel << 3];
        h8 a1 = *(const h8*)&lA[bi][mL][16 + (halfsel << 3)];
        // B 32x16 f16 fragment: same K mapping as A
        h8 b0 = *(const h8*)&lB[bi][nL][halfsel << 3];
        h8 b1 = *(const h8*)&lB[bi][nL][16 + (halfsel << 3)];
        v16h af, bf;
#pragma unroll
        for (int j = 0; j < 8; ++j) {
            af[j] = a0[j]; af[j + 8] = a1[j];
            bf[j] = b0[j]; bf[j + 8] = b1[j];
        }
        acc = __builtin_amdgcn_wmma_f32_16x16x32_f16(false, af, false, bf,
                                                     (short)0, acc, false, false);
    }

    // epilogue: C/D layout: VGPR i -> m = i + 8*half, n = lane&15 (all in range)
    int n = n0 + (wc << 4) + l15;
    float bb = bias[n];
    int mBase = m0 + (wr << 4) + (halfsel << 3);
#pragma unroll
    for (int i = 0; i < 8; ++i) {
        float v = acc[i] + bb;
        if (act == ACT_RELU)       v = v > 0.0f ? v : 0.0f;
        else if (act == ACT_LRELU) v = v > 0.0f ? v : 0.01f * v;
        size_t idx = (size_t)(mBase + i) * ldc + n;
        if (accum) v += C[idx];
        C[idx] = v;
    }
}

// ---------------------------------------------------------------------------
// GENERIC WMMA GEMM (edge sizes: K=36 feature layer, N=31 head outputs).
// Same tiling, single-buffered, clamp+mask edge handling.  M%32==0 required.
// ---------------------------------------------------------------------------
__global__ __launch_bounds__(256)
void gemm_wmma_generic(const float* __restrict__ A, int lda,
                       const float* __restrict__ W, int ldw,
                       const float* __restrict__ bias,
                       float* __restrict__ C, int ldc,
                       int M, int N, int K, int act, int accum)
{
    __shared__ __align__(16) _Float16 lA[32][40];
    __shared__ __align__(16) _Float16 lB[64][40];

    int ntN = (N + 63) >> 6;
    int tm = blockIdx.x / ntN;
    int tn = blockIdx.x - tm * ntN;
    int m0 = tm << 5, n0 = tn << 6;

    int tid  = threadIdx.x;
    int lane = tid & 31;
    int wv   = tid >> 5;
    int wr   = wv >> 2;
    int wc   = wv & 3;
    int halfsel = lane >> 4;
    int l15  = lane & 15;

    int ar = tid >> 3;
    int akq = (tid & 7) << 2;
    int bk = tid >> 3;
    int bn8 = (tid & 7) << 3;

    const float* Arow = A + (size_t)(m0 + ar) * lda;

    int mL = (wr << 4) + l15;
    int nL = (wc << 4) + l15;

    v8f acc = {};
    for (int k0 = 0; k0 < K; k0 += 32) {
        __syncthreads();
#pragma unroll
        for (int c = 0; c < 4; ++c) {
            int k = k0 + akq + c;
            int kc = k < K ? k : K - 1;
            float v = Arow[kc];
            v = k < K ? v : 0.0f;
            lA[ar][akq + c] = (_Float16)v;
        }
        {
            int k = k0 + bk;
            int kc = k < K ? k : K - 1;
            bool kok = k < K;
            const float* Wrow = W + (size_t)kc * ldw;
#pragma unroll
            for (int c = 0; c < 8; ++c) {
                int n = n0 + bn8 + c;
                int nc = n < N ? n : N - 1;
                float v = Wrow[nc];
                v = (kok && n < N) ? v : 0.0f;
                lB[bn8 + c][bk] = (_Float16)v;
            }
        }
        __syncthreads();
        h8 a0 = *(const h8*)&lA[mL][halfsel << 3];
        h8 a1 = *(const h8*)&lA[mL][16 + (halfsel << 3)];
        h8 b0 = *(const h8*)&lB[nL][halfsel << 3];
        h8 b1 = *(const h8*)&lB[nL][16 + (halfsel << 3)];
        v16h af, bf;
#pragma unroll
        for (int j = 0; j < 8; ++j) {
            af[j] = a0[j]; af[j + 8] = a1[j];
            bf[j] = b0[j]; bf[j + 8] = b1[j];
        }
        acc = __builtin_amdgcn_wmma_f32_16x16x32_f16(false, af, false, bf,
                                                     (short)0, acc, false, false);
    }

    int n = n0 + (wc << 4) + l15;
    if (n < N) {
        float bb = bias[n];
        int mBase = m0 + (wr << 4) + (halfsel << 3);
#pragma unroll
        for (int i = 0; i < 8; ++i) {
            float v = acc[i] + bb;
            if (act == ACT_RELU)       v = v > 0.0f ? v : 0.0f;
            else if (act == ACT_LRELU) v = v > 0.0f ? v : 0.01f * v;
            size_t idx = (size_t)(mBase + i) * ldc + n;
            if (accum) v += C[idx];
            C[idx] = v;
        }
    }
}

// ---------------------------------------------------------------------------
// Stage 1: input features  x36[row,36] = [hex_emb | player_emb | idx_emb]
// ---------------------------------------------------------------------------
__global__ void k_features(const int* __restrict__ st, const int* __restrict__ aid,
                           const float* __restrict__ hex_emb,
                           const float* __restrict__ player_emb,
                           const float* __restrict__ idx_emb,
                           float* __restrict__ x36)
{
    int i = blockIdx.x * blockDim.x + threadIdx.x;
    if (i >= CROWS * 36) return;
    int row = i / 36, c = i - row * 36;
    int ba = row / CT;
    int a  = ba % CA;
    float v;
    if (c < 12) {
        int h = st[row];
        h = h < 0 ? 0 : (h > 598 ? 598 : h);
        v = hex_emb[h * 12 + c];
    } else if (c < 24) {
        int p = aid[ba]; if (p < 0) p = 0;
        v = player_emb[p * 12 + (c - 12)];
    } else {
        v = idx_emb[a * 12 + (c - 24)];
    }
    x36[(size_t)row * 36 + c] = v;
}

// per-channel mean/var over all rows (batch-norm, biased variance)
__global__ void k_colstats(const float* __restrict__ z, int Mrows, int Ccols,
                           float* __restrict__ mean, float* __restrict__ var)
{
    int c = blockIdx.x;
    __shared__ float s1[256], s2[256];
    float a = 0.0f, b = 0.0f;
    for (int r = threadIdx.x; r < Mrows; r += 256) {
        float v = z[(size_t)r * Ccols + c];
        a += v; b += v * v;
    }
    s1[threadIdx.x] = a; s2[threadIdx.x] = b;
    __syncthreads();
    for (int s = 128; s > 0; s >>= 1) {
        if (threadIdx.x < (unsigned)s) { s1[threadIdx.x] += s1[threadIdx.x + s]; s2[threadIdx.x] += s2[threadIdx.x + s]; }
        __syncthreads();
    }
    if (threadIdx.x == 0) {
        float mu = s1[0] / (float)Mrows;
        mean[c] = mu;
        var[c]  = s2[0] / (float)Mrows - mu * mu;
    }
}

__global__ void k_bn_relu(const float* __restrict__ z, const float* __restrict__ mean,
                          const float* __restrict__ var, const float* __restrict__ g,
                          const float* __restrict__ b, float* __restrict__ out,
                          int Mrows, int Ccols)
{
    int i = blockIdx.x * blockDim.x + threadIdx.x;
    if (i >= Mrows * Ccols) return;
    int c = i % Ccols;
    float v = (z[i] - mean[c]) * rsqrtf(var[c] + 1e-5f) * g[c] + b[c];
    out[i] = v > 0.0f ? v : 0.0f;
}

// ---------------------------------------------------------------------------
// Stage 2: GATv2 layers
// ---------------------------------------------------------------------------
__global__ void k_to_graph(const float* __restrict__ x, float* __restrict__ xg)
{
    int i = blockIdx.x * blockDim.x + threadIdx.x;
    if (i >= CROWS * CD) return;
    int d = i & 127; int rowg = i >> 7;
    int a = rowg % CA; int g = rowg / CA;
    int b = g / CT, t = g - b * CT;
    xg[i] = x[(size_t)(((b * CA + a) * CT) + t) * CD + d];
}

// one wave per (g,e,h): score = sum_d lrelu0.2(xl[src]+xr[dst]+ea*we+be) * att
__global__ void k_gat_score(const float* __restrict__ xl, const float* __restrict__ xr,
                            const float* __restrict__ ea, const int* __restrict__ eidx,
                            const float* __restrict__ we, const float* __restrict__ be,
                            const float* __restrict__ att, float* __restrict__ score)
{
    int wid = blockIdx.x * (blockDim.x >> 5) + (threadIdx.x >> 5);
    if (wid >= CG * CE * CH) return;
    int h = wid % CH; int t2 = wid / CH;
    int e = t2 % CE;  int g  = t2 / CE;
    int lane = threadIdx.x & 31;
    int s = eidx[e], dd = eidx[CE + e];
    float eav = ea[(size_t)g * CE + e];
    const float* pl = xl + ((size_t)(g * CA + s)  * (CH * CD) + h * CD);
    const float* pr = xr + ((size_t)(g * CA + dd) * (CH * CD) + h * CD);
    const float* pw = we  + h * CD;
    const float* pb = be  + h * CD;
    const float* pa = att + h * CD;
    float acc = 0.0f;
    for (int d = lane; d < CD; d += 32) {
        float z = pl[d] + pr[d] + eav * pw[d] + pb[d];
        z = z > 0.0f ? z : 0.2f * z;
        acc += z * pa[d];
    }
    for (int o = 16; o > 0; o >>= 1) acc += __shfl_xor(acc, o, 32);
    if (lane == 0) score[((size_t)g * CE + e) * CH + h] = acc;
}

// per (g,h,dst): softmax over incoming edges
__global__ void k_gat_alpha(const int* __restrict__ eidx, const float* __restrict__ score,
                            float* __restrict__ alpha)
{
    int i = blockIdx.x * blockDim.x + threadIdx.x;
    if (i >= CG * CH * CA) return;
    int a = i % CA; int t2 = i / CA;
    int h = t2 % CH; int g = t2 / CH;
    const int* dst = eidx + CE;
    float m = -1e30f;
    for (int e = 0; e < CE; ++e)
        if (dst[e] == a) { float s = score[((size_t)g * CE + e) * CH + h]; m = s > m ? s : m; }
    float sum = 0.0f;
    for (int e = 0; e < CE; ++e)
        if (dst[e] == a) {
            float w = expf(score[((size_t)g * CE + e) * CH + h] - m);
            alpha[((size_t)g * CE + e) * CH + h] = w;
            sum += w;
        }
    float inv = 1.0f / sum;
    for (int e = 0; e < CE; ++e)
        if (dst[e] == a) alpha[((size_t)g * CE + e) * CH + h] *= inv;
}

// per (g,dst,d): out = mean_h sum_{e in dst} alpha * xl[src] + bias
__global__ void k_gat_agg(const float* __restrict__ xl, const float* __restrict__ alpha,
                          const int* __restrict__ eidx, const float* __restrict__ gbias,
                          float* __restrict__ yg)
{
    int i = blockIdx.x * blockDim.x + threadIdx.x;
    if (i >= CG * CA * CD) return;
    int d = i & 127; int r = i >> 7;
    int a = r % CA; int g = r / CA;
    const int* src = eidx; const int* dst = eidx + CE;
    float acc = 0.0f;
    for (int e = 0; e < CE; ++e) {
        if (dst[e] == a) {
            int s = src[e];
            const float* px  = xl + ((size_t)(g * CA + s) * (CH * CD) + d);
            const float* pal = alpha + ((size_t)g * CE + e) * CH;
#pragma unroll
            for (int h = 0; h < CH; ++h) acc += pal[h] * px[h * CD];
        }
    }
    yg[i] = acc * 0.125f + gbias[d];
}

// x (BA-major, in place) = LN(x + permute(yg))
__global__ void k_res_ln_graph(float* x, const float* __restrict__ yg,
                               const float* __restrict__ g, const float* __restrict__ b)
{
    int row = blockIdx.x;
    int d = threadIdx.x;
    int ba = row / CT, t = row - ba * CT;
    int bb = ba / CA, a = ba - bb * CA;
    size_t gi = ((size_t)((bb * CT + t) * CA) + a) * CD + d;
    float v = x[(size_t)row * CD + d] + yg[gi];
    __shared__ float s1[128], s2[128];
    s1[d] = v; s2[d] = v * v;
    __syncthreads();
    for (int s = 64; s > 0; s >>= 1) {
        if (d < s) { s1[d] += s1[d + s]; s2[d] += s2[d + s]; }
        __syncthreads();
    }
    float mu = s1[0] * (1.0f / 128.0f);
    float vr = s2[0] * (1.0f / 128.0f) - mu * mu;
    x[(size_t)row * CD + d] = (v - mu) * rsqrtf(vr + 1e-5f) * g[d] + b[d];
}

// plain layernorm over last dim 128 (row per block)
__global__ void k_ln(const float* __restrict__ in, float* __restrict__ out,
                     const float* __restrict__ g, const float* __restrict__ b)
{
    int row = blockIdx.x;
    int d = threadIdx.x;
    float v = in[(size_t)row * CD + d];
    __shared__ float s1[128], s2[128];
    s1[d] = v; s2[d] = v * v;
    __syncthreads();
    for (int s = 64; s > 0; s >>= 1) {
        if (d < s) { s1[d] += s1[d + s]; s2[d] += s2[d + s]; }
        __syncthreads();
    }
    float mu = s1[0] * (1.0f / 128.0f);
    float vr = s2[0] * (1.0f / 128.0f) - mu * mu;
    out[(size_t)row * CD + d] = (v - mu) * rsqrtf(vr + 1e-5f) * g[d] + b[d];
}

// ---------------------------------------------------------------------------
// Stage 3: decoder
// ---------------------------------------------------------------------------
__global__ void k_build_seq(const float* __restrict__ x, const int* __restrict__ atok,
                            const float* __restrict__ ball_emb,
                            const float* __restrict__ pl_emb, float* __restrict__ h)
{
    int i = blockIdx.x * blockDim.x + threadIdx.x;
    if (i >= 12 * CG * CD) return;
    int d = i & 127; int r = i >> 7;
    int n = r % CG; int l = r / CG;
    int b = n / CT, t = n - b * CT;
    float v;
    if ((l & 1) == 0) {
        int pi = l >> 1;
        v = x[(size_t)(((b * CA + pi) * CT) + t) * CD + d];
    } else {
        int pi = (l - 1) >> 1;
        int tok = atok[(b * 6 + pi) * CT + t];
        v = (pi == 0) ? ball_emb[tok * CD + d] : pl_emb[tok * CD + d];
    }
    float div = expf(-(float)(d & ~1) * (9.210340372f / 128.0f));  // ln(10000)/D
    float pe = (d & 1) ? cosf((float)l * div) : sinf((float)l * div);
    h[i] = v + pe;
}

__global__ void k_build_mem(const float* __restrict__ x, float* __restrict__ mem)
{
    int i = blockIdx.x * blockDim.x + threadIdx.x;
    if (i >= CA * CG * CD) return;
    int d = i & 127; int r = i >> 7;
    int n = r % CG; int a = r / CG;
    int b = n / CT, t = n - b * CT;
    mem[i] = x[(size_t)(((b * CA + a) * CT) + t) * CD + d];
}

// causal self-attention, L=12, heads 8, dh=16; qkv rows (l*320+n), cols [q|k|v]
__global__ void k_self_attn(const float* __restrict__ qkv, float* __restrict__ ao)
{
    int i = blockIdx.x * blockDim.x + threadIdx.x;
    if (i >= CG * CH * 12) return;
    int lq = i % 12; int t2 = i / 12;
    int h = t2 % CH; int n = t2 / CH;
    float q[16];
    const float* qp = qkv + ((size_t)(lq * CG + n) * 384 + h * 16);
#pragma unroll
    for (int j = 0; j < 16; ++j) q[j] = qp[j];
    float sc[12]; float m = -1e30f;
    for (int lk = 0; lk <= lq; ++lk) {
        const float* kp = qkv + ((size_t)(lk * CG + n) * 384 + 128 + h * 16);
        float s = 0.0f;
#pragma unroll
        for (int j = 0; j < 16; ++j) s += q[j] * kp[j];
        s *= 0.25f;
        sc[lk] = s; m = s > m ? s : m;
    }
    float sum = 0.0f;
    for (int lk = 0; lk <= lq; ++lk) { sc[lk] = expf(sc[lk] - m); sum += sc[lk]; }
    float inv = 1.0f / sum;
    float o[16];
#pragma unroll
    for (int j = 0; j < 16; ++j) o[j] = 0.0f;
    for (int lk = 0; lk <= lq; ++lk) {
        const float* vp = qkv + ((size_t)(lk * CG + n) * 384 + 256 + h * 16);
        float p = sc[lk] * inv;
#pragma unroll
        for (int j = 0; j < 16; ++j) o[j] += p * vp[j];
    }
    float* op = ao + ((size_t)(lq * CG + n) * CD + h * 16);
#pragma unroll
    for (int j = 0; j < 16; ++j) op[j] = o[j];
}

// cross-attention: q rows (lq*320+n, 128), kv rows (a*320+n, 256 = [k|v])
__global__ void k_cross_attn(const float* __restrict__ qb, const float* __restrict__ kv,
                             float* __restrict__ ao)
{
    int i = blockIdx.x * blockDim.x + threadIdx.x;
    if (i >= CG * CH * 12) return;
    int lq = i % 12; int t2 = i / 12;
    int h = t2 % CH; int n = t2 / CH;
    float q[16];
    const float* qp = qb + ((size_t)(lq * CG + n) * CD + h * 16);
#pragma unroll
    for (int j = 0; j < 16; ++j) q[j] = qp[j];
    float sc[12]; float m = -1e30f;
    for (int a = 0; a < CA; ++a) {
        const float* kp = kv + ((size_t)(a * CG + n) * 256 + h * 16);
        float s = 0.0f;
#pragma unroll
        for (int j = 0; j < 16; ++j) s += q[j] * kp[j];
        s *= 0.25f;
        sc[a] = s; m = s > m ? s : m;
    }
    float sum = 0.0f;
    for (int a = 0; a < CA; ++a) { sc[a] = expf(sc[a] - m); sum += sc[a]; }
    float inv = 1.0f / sum;
    float o[16];
#pragma unroll
    for (int j = 0; j < 16; ++j) o[j] = 0.0f;
    for (int a = 0; a < CA; ++a) {
        const float* vp = kv + ((size_t)(a * CG + n) * 256 + 128 + h * 16);
        float p = sc[a] * inv;
#pragma unroll
        for (int j = 0; j < 16; ++j) o[j] += p * vp[j];
    }
    float* op = ao + ((size_t)(lq * CG + n) * CD + h * 16);
#pragma unroll
    for (int j = 0; j < 16; ++j) op[j] = o[j];
}

// play tokens: pt[(n*5+i),:] = hN[(2+2i)*320+n,:]
__global__ void k_gather_play(const float* __restrict__ hN, float* __restrict__ pt)
{
    int i = blockIdx.x * blockDim.x + threadIdx.x;
    if (i >= CG * 5 * CD) return;
    int d = i & 127; int r = i >> 7;
    int pi = r % 5; int n = r / 5;
    pt[i] = hN[(size_t)((2 + 2 * pi) * CG + n) * CD + d];
}

// ---------------------------------------------------------------------------
// Host orchestration
// ---------------------------------------------------------------------------
static void gemm(hipStream_t s, const float* A, int lda, const float* W, int ldw,
                 const float* bias, float* C, int ldc, int M, int N, int K,
                 int act, int accum)
{
    int blocks = ((M + 31) / 32) * ((N + 63) / 64);
    bool full = ((K & 31) == 0) && ((N & 63) == 0) && ((lda & 3) == 0) && ((ldw & 3) == 0);
    if (full)
        gemm_wmma_full<<<blocks, 256, 0, s>>>(A, lda, W, ldw, bias, C, ldc, M, N, K, act, accum);
    else
        gemm_wmma_generic<<<blocks, 256, 0, s>>>(A, lda, W, ldw, bias, C, ldc, M, N, K, act, accum);
}

extern "C" void kernel_launch(void* const* d_in, const int* in_sizes, int n_in,
                              void* d_out, int out_size, void* d_ws, size_t ws_size,
                              hipStream_t stream)
{
    (void)in_sizes; (void)n_in; (void)out_size; (void)ws_size;
    // ---- inputs (setup_inputs() dict insertion order, params flattened
    //      recursively in insertion order) ----
    const int*   st   = (const int*)d_in[0];   // state_tokens (48,80,1)
    const int*   aid  = (const int*)d_in[1];   // agent_ids (48)
    //            d_in[2] padding_mask: unused
    const int*   atok = (const int*)d_in[3];   // action_tokens (24,80)
    const int*   eidx = (const int*)d_in[4];   // edge_index (2,132)
    const float* eatt = (const float*)d_in[5]; // edge_attr (4,80,132,1)
    auto P = [&](int i) -> const float* { return (const float*)d_in[i]; };

    float* WSF = (float*)d_ws;
    float* X     = WSF;                       // 3840*128
    float* XG    = WSF + 491520;              // 3840*128
    float* XL    = WSF + 983040;              // 3840*1024 region (phase-reused)
    float* XR    = WSF + 4915200;             // 3840*1024 region (phase-reused)
    float* SCORE = WSF + 8847360;             // 320*132*8
    float* ALPHA = WSF + 9185280;             // 320*132*8
    float* YG    = WSF + 9523200;             // 3840*128
    float* H     = WSF + 10014720;            // 3840*128
    float* MEM   = WSF + 10506240;            // 3840*128
    float* PT    = WSF + 10997760;            // 1600*128
    float* T1    = WSF + 11202560;            // 1600*128
    float* MEAN  = WSF + 11407360;            // 256
    float* VAR   = WSF + 11407616;            // 256
    // stage-1 temporaries alias the XL region
    float* X36 = XL;                          // 3840*36
    float* Z1  = XL + 200704;                 // 3840*64
    float* A1  = XL + 500000;                 // 3840*64
    float* Z2  = XL + 800000;                 // 3840*256
    float* A2  = XL + 1800000;                // 3840*256
    float* Z3  = XR;                          // 3840*128
    // stage-3 temporaries alias XL/XR regions
    float* XN  = XR;                          // 3840*128
    float* QKV = XR + 500000;                 // 3840*384
    float* AO  = XR + 2000000;                // 3840*128
    float* KV  = XL;                          // 3840*256
    float* QB  = XL + 1000000;                // 3840*128
    float* HN  = XL + 1500000;                // 3840*128
    float* OUT = (float*)d_out;

    // ---------------- Stage 1: embeddings + BN-MLP ----------------
    k_features<<<(CROWS * 36 + 255) / 256, 256, 0, stream>>>(st, aid, P(6), P(7), P(8), X36);
    gemm(stream, X36, 36, P(9), 64, P(10), Z1, 64, CROWS, 64, 36, ACT_NONE, 0);
    k_colstats<<<64, 256, 0, stream>>>(Z1, CROWS, 64, MEAN, VAR);
    k_bn_relu<<<(CROWS * 64 + 255) / 256, 256, 0, stream>>>(Z1, MEAN, VAR, P(11), P(12), A1, CROWS, 64);
    gemm(stream, A1, 64, P(13), 256, P(14), Z2, 256, CROWS, 256, 64, ACT_NONE, 0);
    k_colstats<<<256, 256, 0, stream>>>(Z2, CROWS, 256, MEAN, VAR);
    k_bn_relu<<<(CROWS * 256 + 255) / 256, 256, 0, stream>>>(Z2, MEAN, VAR, P(15), P(16), A2, CROWS, 256);
    gemm(stream, A2, 256, P(17), 128, P(18), Z3, 128, CROWS, 128, 256, ACT_NONE, 0);
    k_colstats<<<128, 256, 0, stream>>>(Z3, CROWS, 128, MEAN, VAR);
    k_bn_relu<<<(CROWS * 128 + 255) / 256, 256, 0, stream>>>(Z3, MEAN, VAR, P(19), P(20), X, CROWS, 128);

    // ---------------- Stage 2: 3x GATv2 + residual LN ----------------
    for (int L = 0; L < 3; ++L) {
        int g0 = 21 + 8 * L;
        k_to_graph<<<(CROWS * CD + 255) / 256, 256, 0, stream>>>(X, XG);
        gemm(stream, XG, 128, P(g0 + 0), 1024, P(g0 + 1), XL, 1024, CROWS, 1024, 128, ACT_NONE, 0);
        gemm(stream, XG, 128, P(g0 + 2), 1024, P(g0 + 3), XR, 1024, CROWS, 1024, 128, ACT_NONE, 0);
        k_gat_score<<<(CG * CE * CH + 7) / 8, 256, 0, stream>>>(XL, XR, eatt, eidx,
                                                                P(g0 + 4), P(g0 + 5), P(g0 + 6), SCORE);
        k_gat_alpha<<<(CG * CH * CA + 255) / 256, 256, 0, stream>>>(eidx, SCORE, ALPHA);
        k_gat_agg<<<(CG * CA * CD + 255) / 256, 256, 0, stream>>>(XL, ALPHA, eidx, P(g0 + 7), YG);
        k_res_ln_graph<<<CROWS, 128, 0, stream>>>(X, YG, P(45 + 2 * L), P(46 + 2 * L));
    }

    // ---------------- Stage 3: decoder ----------------
    k_build_seq<<<(12 * CG * CD + 255) / 256, 256, 0, stream>>>(X, atok, P(51), P(52), H);
    k_build_mem<<<(CA * CG * CD + 255) / 256, 256, 0, stream>>>(X, MEM);
    for (int k = 0; k < 6; ++k) {
        int b = 53 + 18 * k;
        // self-attention
        k_ln<<<CROWS, 128, 0, stream>>>(H, XN, P(b + 0), P(b + 1));
        gemm(stream, XN, 128, P(b + 2), 384, P(b + 3), QKV, 384, CROWS, 384, 128, ACT_NONE, 0);
        k_self_attn<<<(CG * CH * 12 + 255) / 256, 256, 0, stream>>>(QKV, AO);
        gemm(stream, AO, 128, P(b + 4), 128, P(b + 5), H, 128, CROWS, 128, 128, ACT_NONE, 1);
        // cross-attention (kv from memory, weights are in_w[:, D:3D])
        k_ln<<<CROWS, 128, 0, stream>>>(H, XN, P(b + 6), P(b + 7));
        gemm(stream, XN, 128, P(b + 8), 384, P(b + 9), QB, 128, CROWS, 128, 128, ACT_NONE, 0);
        gemm(stream, MEM, 128, P(b + 8) + 128, 384, P(b + 9) + 128, KV, 256, CROWS, 256, 128, ACT_NONE, 0);
        k_cross_attn<<<(CG * CH * 12 + 255) / 256, 256, 0, stream>>>(QB, KV, AO);
        gemm(stream, AO, 128, P(b + 10), 128, P(b + 11), H, 128, CROWS, 128, 128, ACT_NONE, 1);
        // feed-forward
        k_ln<<<CROWS, 128, 0, stream>>>(H, XN, P(b + 12), P(b + 13));
        gemm(stream, XN, 128, P(b + 14), 128, P(b + 15), AO, 128, CROWS, 128, 128, ACT_LRELU, 0);
        gemm(stream, AO, 128, P(b + 16), 128, P(b + 17), H, 128, CROWS, 128, 128, ACT_NONE, 1);
    }
    k_ln<<<CROWS, 128, 0, stream>>>(H, HN, P(161), P(162));

    // ---------------- Heads ----------------
    gemm(stream, HN, 128, P(171), 128, P(172), T1, 128, CG, 128, 128, ACT_LRELU, 0);
    gemm(stream, T1, 128, P(173), 31, P(174), OUT + 0, 31, CG, 31, 128, ACT_NONE, 0);        // qb1
    gemm(stream, HN, 128, P(175), 128, P(176), T1, 128, CG, 128, 128, ACT_LRELU, 0);
    gemm(stream, T1, 128, P(177), 31, P(178), OUT + 9920, 31, CG, 31, 128, ACT_NONE, 0);     // qb2
    k_gather_play<<<(CG * 5 * CD + 255) / 256, 256, 0, stream>>>(HN, PT);
    gemm(stream, PT, 128, P(163), 128, P(164), T1, 128, CG * 5, 128, 128, ACT_LRELU, 0);
    gemm(stream, T1, 128, P(165), 31, P(166), OUT + 19840, 31, CG * 5, 31, 128, ACT_NONE, 0); // q1
    gemm(stream, PT, 128, P(167), 128, P(168), T1, 128, CG * 5, 128, 128, ACT_LRELU, 0);
    gemm(stream, T1, 128, P(169), 31, P(170), OUT + 69440, 31, CG * 5, 31, 128, ACT_NONE, 0); // q2
}
